// ConformerBlock_82420422410345
// MI455X (gfx1250) — matrix operand order repaired and checked by reading, not compiled, not run on hardware
//
#include <hip/hip_runtime.h>
#include <hip/hip_bf16.h>
#include <math.h>

typedef _Float16 half_t;
typedef __attribute__((ext_vector_type(16))) _Float16 v16h;
typedef __attribute__((ext_vector_type(8)))  _Float16 v8h;
typedef __attribute__((ext_vector_type(8)))  float    v8f;

#define DEV __device__ __forceinline__

DEV v8f wmma_f16(v16h a, v16h b, v8f c) {
  // D = A(16x32 f16) * B(32x16 f16) + C(16x16 f32)
  return __builtin_amdgcn_wmma_f32_16x16x32_f16(false, a, false, b, (short)0, c,
                                                false, false);
}

// A-fragment from LDS tile (row-major, LD halfs per row), per ISA 7.12.2:
// lane<16: K in {c0..c0+7, c0+16..c0+23}; lane>=16: +8.
template <int LD>
DEV v16h frag_a(const half_t* base, int row0, int col0, int lane) {
  const int hs = (lane >> 4) * 8, r = lane & 15;
  const half_t* p = base + (size_t)(row0 + r) * LD + col0;
  v8h lo = *(const v8h*)(p + hs);
  v8h hi = *(const v8h*)(p + 16 + hs);
  return __builtin_shufflevector(lo, hi, 0, 1, 2, 3, 4, 5, 6, 7, 8, 9, 10, 11,
                                 12, 13, 14, 15);
}

// B-fragment from transposed tile BT[n][k]: lane<16 holds K in
// {c0..c0+7, c0+16..c0+23} for column n=lane; lane>=16: +8.
template <int LD>
DEV v16h frag_b(const half_t* base, int row0, int col0, int lane) {
  const int hs = (lane >> 4) * 8, r = lane & 15;
  const half_t* p = base + (size_t)(row0 + r) * LD + col0;
  v8h lo = *(const v8h*)(p + hs);
  v8h hi = *(const v8h*)(p + 16 + hs);
  return __builtin_shufflevector(lo, hi, 0, 1, 2, 3, 4, 5, 6, 7, 8, 9, 10, 11,
                                 12, 13, 14, 15);
}

// ---------------------------------------------------------------------------
// Generic GEMM: out[M,N] = epilogue(A[M,K] @ W[K,N] + bias)
// A: f16 row-major (ld=K); BT: f16 = W^T row-major [N][K] (ld=K).
// MODE 0: outh = (half)(alpha*(acc+bias))
// MODE 1: outh = (half)swish(acc+bias)
// MODE 2: outf = alpha*(acc+bias) + res
// ---------------------------------------------------------------------------
template <int MODE>
__global__ __launch_bounds__(256) void gemm_kernel(
    const half_t* __restrict__ A, const half_t* __restrict__ BT,
    const float* __restrict__ bias, half_t* __restrict__ outh,
    float* __restrict__ outf, const float* __restrict__ res, int N, int K,
    float alpha) {
  __shared__ alignas(32) half_t As[128][32];
  __shared__ alignas(32) half_t Bs[128][32];
  const int tid = threadIdx.x, lane = tid & 31, wave = tid >> 5;
  const int wm = wave & 3, wn = wave >> 2;  // 4 x 2 wave grid
  const int m0 = blockIdx.y * 128, n0 = blockIdx.x * 128;
  const v8f vzero = {0.f, 0.f, 0.f, 0.f, 0.f, 0.f, 0.f, 0.f};
  v8f acc[2][4];
#pragma unroll
  for (int mi = 0; mi < 2; mi++)
#pragma unroll
    for (int ni = 0; ni < 4; ni++) acc[mi][ni] = vzero;

  const int nk = K >> 5;
  for (int kt = 0; kt < nk; ++kt) {
    const int kb = kt << 5;
#pragma unroll
    for (int rr = 0; rr < 2; ++rr) {
      int c = tid + rr * 256;                 // 512 chunks of 8 halves
      int row = c >> 2, col = (c & 3) << 3;
      *(uint4*)&As[row][col] =
          *(const uint4*)(A + (size_t)(m0 + row) * K + kb + col);
      *(uint4*)&Bs[row][col] =
          *(const uint4*)(BT + (size_t)(n0 + row) * K + kb + col);
    }
    __syncthreads();
    v16h af[2], bf[4];
#pragma unroll
    for (int mi = 0; mi < 2; mi++)
      af[mi] = frag_a<32>(&As[0][0], wm * 32 + mi * 16, 0, lane);
#pragma unroll
    for (int ni = 0; ni < 4; ni++)
      bf[ni] = frag_b<32>(&Bs[0][0], wn * 64 + ni * 16, 0, lane);
#pragma unroll
    for (int mi = 0; mi < 2; mi++)
#pragma unroll
      for (int ni = 0; ni < 4; ni++)
        acc[mi][ni] = wmma_f16(af[mi], bf[ni], acc[mi][ni]);
    __syncthreads();
  }

  const int hs = (lane >> 4) * 8, l15 = lane & 15;
#pragma unroll
  for (int mi = 0; mi < 2; mi++) {
#pragma unroll
    for (int ni = 0; ni < 4; ni++) {
      const int colg = n0 + wn * 64 + ni * 16 + l15;
      const float bc = bias[colg];
#pragma unroll
      for (int v = 0; v < 8; v++) {
        const int rowg = m0 + wm * 32 + mi * 16 + hs + v;
        float hv = acc[mi][ni][v] + bc;
        const size_t idx = (size_t)rowg * N + colg;
        if (MODE == 0)
          outh[idx] = (half_t)(alpha * hv);
        else if (MODE == 1)
          outh[idx] = (half_t)(hv / (1.f + __expf(-hv)));  // swish
        else
          outf[idx] = alpha * hv + res[idx];
      }
    }
  }
}

// ---------------------------------------------------------------------------
// LayerNorm over last dim (512): one block per row, 256 threads x 2 elems.
// ---------------------------------------------------------------------------
__global__ __launch_bounds__(256) void ln_kernel(
    const float* __restrict__ x, const float* __restrict__ g,
    const float* __restrict__ b, half_t* __restrict__ o16,
    float* __restrict__ o32, int write_f32) {
  __shared__ float sred[8], ssred[8];
  const int row = blockIdx.x, tid = threadIdx.x;
  const float* xr = x + (size_t)row * 512;
  float v0 = xr[tid], v1 = xr[tid + 256];
  float s = v0 + v1, ss = v0 * v0 + v1 * v1;
#pragma unroll
  for (int m = 16; m >= 1; m >>= 1) {
    s += __shfl_xor(s, m, 32);
    ss += __shfl_xor(ss, m, 32);
  }
  if ((tid & 31) == 0) {
    sred[tid >> 5] = s;
    ssred[tid >> 5] = ss;
  }
  __syncthreads();
  float ts = 0.f, tss = 0.f;
#pragma unroll
  for (int i = 0; i < 8; i++) {
    ts += sred[i];
    tss += ssred[i];
  }
  const float mean = ts * (1.f / 512.f);
  const float var = tss * (1.f / 512.f) - mean * mean;
  const float rstd = rsqrtf(var + 1e-5f);
  const float y0 = (v0 - mean) * rstd * g[tid] + b[tid];
  const float y1 = (v1 - mean) * rstd * g[tid + 256] + b[tid + 256];
  if (write_f32) {
    o32[(size_t)row * 512 + tid] = y0;
    o32[(size_t)row * 512 + tid + 256] = y1;
  } else {
    o16[(size_t)row * 512 + tid] = (half_t)y0;
    o16[(size_t)row * 512 + tid + 256] = (half_t)y1;
  }
}

// Weight convert + transpose: W[K][N] f32 -> WT[N][K] f16
__global__ void wtrans_kernel(const float* __restrict__ W,
                              half_t* __restrict__ WT, int K, int N) {
  int gid = blockIdx.x * 256 + threadIdx.x;
  if (gid >= K * N) return;
  int k = gid / N, n = gid - k * N;
  WT[(size_t)n * K + k] = (half_t)W[gid];
}

__global__ void f32to16_kernel(const float* __restrict__ X,
                               half_t* __restrict__ Y, int n) {
  int gid = blockIdx.x * 256 + threadIdx.x;
  if (gid < n) Y[gid] = (half_t)X[gid];
}

// p[bh,i,t] = q_scaled[b,i,h,:] . rel_emb[t,:]   (t in 0..62)
__global__ void relp_kernel(const half_t* __restrict__ QB,
                            const float* __restrict__ rel,
                            float* __restrict__ PB) {
  int gid = blockIdx.x * 256 + threadIdx.x;
  if (gid >= 32768 * 63) return;
  int row = gid / 63, t = gid - row * 63;
  int bh = row >> 9, i = row & 511;
  int b = bh >> 3, h = bh & 7;
  const half_t* q = QB + (size_t)(b * 512 + i) * 512 + h * 64;
  const float* r = rel + t * 64;
  float acc = 0.f;
#pragma unroll
  for (int d = 0; d < 64; d++) acc += (float)q[d] * r[d];
  PB[(size_t)row * 64 + t] = acc;
}

// GLU over channels: out[row,c] = a * sigmoid(g), a=H[:, :1024], g=H[:,1024:]
__global__ void glu_kernel(const half_t* __restrict__ H,
                           half_t* __restrict__ O) {
  int gid = blockIdx.x * 256 + threadIdx.x;  // 4096*1024 total (exact)
  int row = gid >> 10, c = gid & 1023;
  float a = (float)H[(size_t)row * 2048 + c];
  float g = (float)H[(size_t)row * 2048 + 1024 + c];
  O[gid] = (half_t)(a / (1.f + __expf(-g)));
}

// Causal depthwise conv (K=31) + BN(eval) + swish. X/Y: f16 [B*N][1024].
__global__ __launch_bounds__(256) void dwconv_kernel(
    const half_t* __restrict__ X, const float* __restrict__ W,
    const float* __restrict__ bias, const float* __restrict__ bng,
    const float* __restrict__ bnb, const float* __restrict__ bnm,
    const float* __restrict__ bnv, half_t* __restrict__ Y) {
  const int c = blockIdx.x * 256 + threadIdx.x;  // channel
  const int n0 = blockIdx.y * 128;
  const int b = blockIdx.z;
  float w[31];
#pragma unroll
  for (int t = 0; t < 31; t++) w[t] = W[c * 31 + t];
  const float scale = rsqrtf(bnv[c] + 1e-5f) * bng[c];
  const float shift = bnb[c] - bnm[c] * scale;
  const float db = bias[c];
  float win[31];
#pragma unroll
  for (int t = 0; t < 31; t++) {
    int n = n0 - 31 + t;
    win[t] = (n >= 0) ? (float)X[(size_t)(b * 512 + n) * 1024 + c] : 0.f;
  }
  for (int n = n0; n < n0 + 128; ++n) {
#pragma unroll
    for (int t = 0; t < 30; t++) win[t] = win[t + 1];
    win[30] = (float)X[(size_t)(b * 512 + n) * 1024 + c];
    float acc = db;
#pragma unroll
    for (int t = 0; t < 31; t++) acc += w[t] * win[t];
    float y = acc * scale + shift;
    y = y / (1.f + __expf(-y));
    Y[(size_t)(b * 512 + n) * 1024 + c] = (half_t)y;
  }
}

// ---------------------------------------------------------------------------
// Flash attention per (bh, 128-query tile). Scale folded into Q.
// S = Qs . Ks^T (+ rel p-table), online softmax, O += P . V.
// ---------------------------------------------------------------------------
__global__ __launch_bounds__(256) void attn_kernel(
    const half_t* __restrict__ QB, const half_t* __restrict__ KVB,
    const float* __restrict__ PB, half_t* __restrict__ OB) {
  __shared__ alignas(32) half_t qs[128][64];
  __shared__ alignas(32) half_t ks[64][64];
  __shared__ alignas(32) half_t vs[64][64];  // transposed: vs[d][j]
  __shared__ alignas(32) half_t ps[128][64];
  __shared__ half_t prl[128][63];
  const int tid = threadIdx.x, lane = tid & 31, wave = tid >> 5;
  const int bh = blockIdx.y, b = bh >> 3, h = bh & 7;
  const int i0 = blockIdx.x * 128;

#pragma unroll
  for (int rr = 0; rr < 4; rr++) {
    int c = tid + rr * 256, row = c >> 3, col = (c & 7) << 3;
    *(uint4*)&qs[row][col] =
        *(const uint4*)(QB + (size_t)(b * 512 + i0 + row) * 512 + h * 64 + col);
  }
  for (int c = tid; c < 128 * 63; c += 256) {
    int row = c / 63, t = c - row * 63;
    prl[row][t] = (half_t)PB[(size_t)(bh * 512 + i0 + row) * 64 + t];
  }

  const int r0 = wave * 16;                    // wave owns 16 query rows
  const int hs = (lane >> 4) * 8, l15 = lane & 15;
  const v8f vzero = {0.f, 0.f, 0.f, 0.f, 0.f, 0.f, 0.f, 0.f};
  v8f oacc[4];
#pragma unroll
  for (int ni = 0; ni < 4; ni++) oacc[ni] = vzero;
  float mrun[8], lrun[8];
#pragma unroll
  for (int v = 0; v < 8; v++) {
    mrun[v] = -1e30f;
    lrun[v] = 0.f;
  }

  for (int jt = 0; jt < 8; ++jt) {
    const int j0 = jt * 64;
    __syncthreads();  // also covers qs/prl readiness on first iteration
#pragma unroll
    for (int rr = 0; rr < 2; rr++) {
      int c = tid + rr * 256, row = c >> 3, col = (c & 7) << 3;
      *(uint4*)&ks[row][col] = *(const uint4*)(
          KVB + (size_t)(b * 512 + j0 + row) * 1024 + h * 64 + col);
      uint4 raw = *(const uint4*)(KVB + (size_t)(b * 512 + j0 + row) * 1024 +
                                  512 + h * 64 + col);
      union { uint4 u; half_t e[8]; } uu;
      uu.u = raw;
#pragma unroll
      for (int u = 0; u < 8; u++) vs[col + u][row] = uu.e[u];
    }
    __syncthreads();

    v8f s[4];
#pragma unroll
    for (int ni = 0; ni < 4; ni++) s[ni] = vzero;
#pragma unroll
    for (int kk = 0; kk < 2; kk++) {
      v16h aq = frag_a<64>(&qs[0][0], r0, kk * 32, lane);
#pragma unroll
      for (int ni = 0; ni < 4; ni++) {
        v16h bk = frag_b<64>(&ks[0][0], ni * 16, kk * 32, lane);
        s[ni] = wmma_f16(aq, bk, s[ni]);
      }
    }

#pragma unroll
    for (int v = 0; v < 8; v++) {
      const int il = r0 + hs + v;
      const int ig = i0 + il;
      float rowm = -1e30f;
#pragma unroll
      for (int ni = 0; ni < 4; ni++) {
        int jg = j0 + ni * 16 + l15;
        int d = ig - jg;
        d = d < -31 ? -31 : (d > 31 ? 31 : d);
        float sv = s[ni][v] + (float)prl[il][d + 31];
        s[ni][v] = sv;
        rowm = fmaxf(rowm, sv);
      }
#pragma unroll
      for (int mk = 1; mk < 16; mk <<= 1)
        rowm = fmaxf(rowm, __shfl_xor(rowm, mk, 32));
      const float mnew = fmaxf(mrun[v], rowm);
      const float corr = __expf(mrun[v] - mnew);
      float rs = 0.f;
#pragma unroll
      for (int ni = 0; ni < 4; ni++) {
        float p = __expf(s[ni][v] - mnew);
        s[ni][v] = p;
        rs += p;
      }
#pragma unroll
      for (int mk = 1; mk < 16; mk <<= 1) rs += __shfl_xor(rs, mk, 32);
      lrun[v] = lrun[v] * corr + rs;
      mrun[v] = mnew;
#pragma unroll
      for (int ni = 0; ni < 4; ni++) {
        oacc[ni][v] *= corr;
        ps[il][ni * 16 + l15] = (half_t)s[ni][v];  // wave-local rows only
      }
    }

#pragma unroll
    for (int kk = 0; kk < 2; kk++) {
      v16h ap = frag_a<64>(&ps[0][0], r0, kk * 32, lane);
#pragma unroll
      for (int ni = 0; ni < 4; ni++) {
        v16h bv = frag_b<64>(&vs[0][0], ni * 16, kk * 32, lane);
        oacc[ni] = wmma_f16(ap, bv, oacc[ni]);
      }
    }
  }

#pragma unroll
  for (int ni = 0; ni < 4; ni++)
#pragma unroll
    for (int v = 0; v < 8; v++) {
      float o = oacc[ni][v] / lrun[v];
      OB[(size_t)(b * 512 + i0 + r0 + hs + v) * 512 + h * 64 + ni * 16 + l15] =
          (half_t)o;
    }
}

// ---------------------------------------------------------------------------
extern "C" void kernel_launch(void* const* d_in, const int* in_sizes, int n_in,
                              void* d_out, int out_size, void* d_ws,
                              size_t ws_size, hipStream_t stream) {
  (void)in_sizes; (void)n_in; (void)out_size; (void)ws_size;
  const float* x        = (const float*)d_in[0];
  const float* ff1_ln_g = (const float*)d_in[1];
  const float* ff1_ln_b = (const float*)d_in[2];
  const float* ff1_w1   = (const float*)d_in[3];
  const float* ff1_b1   = (const float*)d_in[4];
  const float* ff1_w2   = (const float*)d_in[5];
  const float* ff1_b2   = (const float*)d_in[6];
  const float* attn_ln_g= (const float*)d_in[7];
  const float* attn_ln_b= (const float*)d_in[8];
  const float* wq       = (const float*)d_in[9];
  const float* bq       = (const float*)d_in[10];
  const float* wkv      = (const float*)d_in[11];
  const float* bkv      = (const float*)d_in[12];
  const float* wo       = (const float*)d_in[13];
  const float* bo       = (const float*)d_in[14];
  const float* rel_emb  = (const float*)d_in[15];
  const float* conv1_w  = (const float*)d_in[16];
  const float* conv1_b  = (const float*)d_in[17];
  const float* dw_w     = (const float*)d_in[18];
  const float* dw_b     = (const float*)d_in[19];
  const float* bn_g     = (const float*)d_in[20];
  const float* bn_b     = (const float*)d_in[21];
  const float* bn_mean  = (const float*)d_in[22];
  const float* bn_var   = (const float*)d_in[23];
  const float* conv2_w  = (const float*)d_in[24];
  const float* conv2_b  = (const float*)d_in[25];
  const float* ff2_ln_g = (const float*)d_in[26];
  const float* ff2_ln_b = (const float*)d_in[27];
  const float* ff2_w1   = (const float*)d_in[28];
  const float* ff2_b1   = (const float*)d_in[29];
  const float* ff2_w2   = (const float*)d_in[30];
  const float* ff2_b2   = (const float*)d_in[31];
  const float* pn_g     = (const float*)d_in[32];
  const float* pn_b     = (const float*)d_in[33];

  char* base = (char*)d_ws;
  size_t off = 0;
  auto take = [&](size_t bytes) -> void* {
    void* p = base + off;
    off = (off + bytes + 255) & ~(size_t)255;
    return p;
  };
  half_t* WT_f1w1 = (half_t*)take((size_t)2048 * 512 * 2);
  half_t* WT_f1w2 = (half_t*)take((size_t)512 * 2048 * 2);
  half_t* WT_q    = (half_t*)take((size_t)512 * 512 * 2);
  half_t* WT_kv   = (half_t*)take((size_t)1024 * 512 * 2);
  half_t* WT_o    = (half_t*)take((size_t)512 * 512 * 2);
  half_t* WT_c1   = (half_t*)take((size_t)2048 * 512 * 2);
  half_t* WT_c2   = (half_t*)take((size_t)512 * 1024 * 2);
  half_t* WT_f2w1 = (half_t*)take((size_t)2048 * 512 * 2);
  half_t* WT_f2w2 = (half_t*)take((size_t)512 * 2048 * 2);
  float*  XA    = (float*)take((size_t)4096 * 512 * 4);
  half_t* LN16  = (half_t*)take((size_t)4096 * 512 * 2);
  half_t* H1    = (half_t*)take((size_t)4096 * 2048 * 2);
  half_t* QB    = (half_t*)take((size_t)4096 * 512 * 2);
  half_t* KVB   = (half_t*)take((size_t)4096 * 1024 * 2);
  float*  PB    = (float*)take((size_t)32768 * 64 * 4);
  half_t* OB    = (half_t*)take((size_t)4096 * 512 * 2);
  half_t* X16   = (half_t*)take((size_t)4096 * 512 * 2);
  half_t* GLU16 = (half_t*)take((size_t)4096 * 1024 * 2);
  half_t* DW16  = (half_t*)take((size_t)4096 * 1024 * 2);

  auto WT = [&](const float* W, half_t* D, int K, int N) {
    int tot = K * N;
    wtrans_kernel<<<(tot + 255) / 256, 256, 0, stream>>>(W, D, K, N);
  };
  WT(ff1_w1, WT_f1w1, 512, 2048);
  WT(ff1_w2, WT_f1w2, 2048, 512);
  WT(wq, WT_q, 512, 512);
  WT(wkv, WT_kv, 512, 1024);
  WT(wo, WT_o, 512, 512);
  WT(conv1_w, WT_c1, 512, 2048);
  WT(conv2_w, WT_c2, 1024, 512);
  WT(ff2_w1, WT_f2w1, 512, 2048);
  WT(ff2_w2, WT_f2w2, 2048, 512);

  const dim3 blk(256);
  // ---- ff1: x = 0.5*FF(LN(x)) + x
  ln_kernel<<<4096, blk, 0, stream>>>(x, ff1_ln_g, ff1_ln_b, LN16, nullptr, 0);
  gemm_kernel<1><<<dim3(16, 32), blk, 0, stream>>>(LN16, WT_f1w1, ff1_b1, H1,
                                                   nullptr, nullptr, 2048, 512, 1.f);
  gemm_kernel<2><<<dim3(4, 32), blk, 0, stream>>>(H1, WT_f1w2, ff1_b2, nullptr,
                                                  XA, x, 512, 2048, 0.5f);
  // ---- attention (scale 1/sqrt(64)=0.125 folded into Q)
  ln_kernel<<<4096, blk, 0, stream>>>(XA, attn_ln_g, attn_ln_b, LN16, nullptr, 0);
  gemm_kernel<0><<<dim3(4, 32), blk, 0, stream>>>(LN16, WT_q, bq, QB, nullptr,
                                                  nullptr, 512, 512, 0.125f);
  gemm_kernel<0><<<dim3(8, 32), blk, 0, stream>>>(LN16, WT_kv, bkv, KVB, nullptr,
                                                  nullptr, 1024, 512, 1.f);
  relp_kernel<<<(32768 * 63 + 255) / 256, 256, 0, stream>>>(QB, rel_emb, PB);
  attn_kernel<<<dim3(4, 64), blk, 0, stream>>>(QB, KVB, PB, OB);
  gemm_kernel<2><<<dim3(4, 32), blk, 0, stream>>>(OB, WT_o, bo, nullptr, XA, XA,
                                                  512, 512, 1.f);
  // ---- conv module
  f32to16_kernel<<<(4096 * 512) / 256, 256, 0, stream>>>(XA, X16, 4096 * 512);
  gemm_kernel<0><<<dim3(16, 32), blk, 0, stream>>>(X16, WT_c1, conv1_b, H1,
                                                   nullptr, nullptr, 2048, 512, 1.f);
  glu_kernel<<<(4096 * 1024) / 256, 256, 0, stream>>>(H1, GLU16);
  dwconv_kernel<<<dim3(4, 4, 8), blk, 0, stream>>>(GLU16, dw_w, dw_b, bn_g, bn_b,
                                                   bn_mean, bn_var, DW16);
  gemm_kernel<2><<<dim3(4, 32), blk, 0, stream>>>(DW16, WT_c2, conv2_b, nullptr,
                                                  XA, XA, 512, 1024, 1.f);
  // ---- ff2 + post-norm
  ln_kernel<<<4096, blk, 0, stream>>>(XA, ff2_ln_g, ff2_ln_b, LN16, nullptr, 0);
  gemm_kernel<1><<<dim3(16, 32), blk, 0, stream>>>(LN16, WT_f2w1, ff2_b1, H1,
                                                   nullptr, nullptr, 2048, 512, 1.f);
  gemm_kernel<2><<<dim3(4, 32), blk, 0, stream>>>(H1, WT_f2w2, ff2_b2, nullptr,
                                                  XA, XA, 512, 2048, 0.5f);
  ln_kernel<<<4096, blk, 0, stream>>>(XA, pn_g, pn_b, nullptr, (float*)d_out, 1);
}
